// Attention_37804302139565
// MI455X (gfx1250) — matrix operand order repaired and checked by reading, not compiled, not run on hardware
//
#include <hip/hip_runtime.h>

// ---------------- problem constants (from reference) ----------------
constexpr int B_   = 2;
constexpr int S_   = 2048;
constexpr int D_   = 2048;
constexpr int H_   = 32;
constexpr int KVH_ = 8;
constexpr int HD_  = 64;         // head dim
constexpr int BS_  = B_ * S_;    // 4096 rows for the projection GEMMs
constexpr int KVD_ = KVH_ * HD_; // 512

typedef __attribute__((ext_vector_type(16))) _Float16     v16h;
typedef __attribute__((ext_vector_type(8)))  float        v8f;
typedef __attribute__((ext_vector_type(4)))  unsigned int u32x4v;
typedef __attribute__((ext_vector_type(4)))  int          i32x4v;
typedef __attribute__((ext_vector_type(8)))  int          i32x8v;

union AFrag { unsigned int u[8]; u32x4v q[2]; v16h v; };

__device__ inline v8f wmma32(const AFrag& a, const AFrag& b, v8f c) {
  // D = A(16x32 f16) * B(32x16 f16) + C(16x16 f32)
  return __builtin_amdgcn_wmma_f32_16x16x32_f16(
      /*neg_a=*/false, a.v, /*neg_b=*/false, b.v,
      /*c_mod=*/(short)0, c, /*reuse_a=*/false, /*reuse_b=*/false);
}

// ---------------- Tensor Data Mover (async 2D tile -> LDS) ------------------
#if __has_builtin(__builtin_amdgcn_tensor_load_to_lds) && \
    __has_builtin(__builtin_amdgcn_s_wait_tensorcnt)
#define USE_TDM 1
#else
#define USE_TDM 0
#endif

#if USE_TDM
// Load a (rows x 32) f16 tile from a row-major (tensor_rows x tensor_k) matrix
// into LDS with a 4-DWORD pad after every 16 DWORDs (row stride 20 dwords).
__device__ inline void tdm_load_tile(unsigned lds_off, const void* gsrc,
                                     int rows, int tensor_k, int tensor_rows) {
  unsigned long long ga = (unsigned long long)(uintptr_t)gsrc;
  u32x4v g0;
  g0[0] = 1u;                                   // count=1, user descriptor
  g0[1] = lds_off;                              // lds_addr [63:32]
  g0[2] = (unsigned)ga;                         // global_addr[31:0] @ bits 95:64
  g0[3] = (unsigned)((ga >> 32) & 0x01FFFFFFull) | (2u << 30); // addr[56:32], type=2
  i32x8v g1 = {};
  g1[0] = (1 << 16)     // data_size = 2 bytes
        | (1 << 20)     // pad_enable
        | (3 << 22)     // pad_interval: 16 DWORDs
        | (3 << 25);    // pad_amount: 4 DWORDs
  g1[1] = (tensor_k & 0xFFFF) << 16;                                   // dim0 lo16
  g1[2] = (int)(((unsigned)tensor_k >> 16) | ((tensor_rows & 0xFFFF) << 16));
  g1[3] = (int)((((unsigned)tensor_rows >> 16) & 0xFFFF) | (32u << 16)); // tile_dim0=32
  g1[4] = rows & 0xFFFF;                                               // tile_dim1
  g1[5] = tensor_k;                                                    // dim0_stride lo32
  i32x4v z4 = {};
#if __has_include(<hip/amd_detail/amd_gfx1250_TDM.h>)
  i32x8v z8 = {};
  __builtin_amdgcn_tensor_load_to_lds(g0, g1, z4, z4, z8, 0);   // clang-23 arity
#else
  __builtin_amdgcn_tensor_load_to_lds(g0, g1, z4, z4, 0);       // ROCm 7.2 arity
#endif
}
#endif

// ---------------- fp32 -> f16 convert (4 elems/thread) ----------------------
__global__ void cvt_kernel(const float* __restrict__ src, _Float16* __restrict__ dst, int n4) {
  int i = blockIdx.x * 256 + threadIdx.x;
  if (i < n4) {
    float4 f = ((const float4*)src)[i];
    union { _Float16 h[4]; unsigned long long u; } p;
    p.h[0] = (_Float16)f.x; p.h[1] = (_Float16)f.y;
    p.h[2] = (_Float16)f.z; p.h[3] = (_Float16)f.w;
    ((unsigned long long*)dst)[i] = p.u;
  }
}

// ---------------- GEMM: C(M,N) = A(M,K) * W(N,K)^T, f16 in, f32 out ---------
// 256 threads (8 waves); block tile 128(M) x 128(N); K step 32; double-buffered
// LDS fed by the Tensor Data Mover (wave 0 drives the DMA engine).
// Wave grid 4(m) x 2(n): each wave owns a 32x64 tile = 2x4 WMMA fragments.
__global__ void __launch_bounds__(256) gemm_f16_nt(
    const _Float16* __restrict__ A, const _Float16* __restrict__ W,
    float* __restrict__ C, int M, int N, int K)
{
  __shared__ unsigned int As[2][128][20]; // 16 dwords data + 4 pad (TDM pad cfg)
  __shared__ unsigned int Bs[2][128][20];
  const int tid   = threadIdx.x;
  const int wave  = tid >> 5;
  const int lane  = tid & 31;
  const int half_ = lane >> 4;
  const int lr    = lane & 15;
  const int wm    = wave & 3;   // 0..3 -> 32-row slice
  const int wn    = wave >> 2;  // 0..1 -> 64-col slice
  const int m0    = blockIdx.y * 128;
  const int n0    = blockIdx.x * 128;

  v8f acc[2][4] = {};

#if USE_TDM
  if (wave == 0) {   // prime buffer 0
    tdm_load_tile((unsigned)(uintptr_t)&As[0][0][0], A + (size_t)m0 * K, 128, K, M);
    tdm_load_tile((unsigned)(uintptr_t)&Bs[0][0][0], W + (size_t)n0 * K, 128, K, N);
  }
#endif

  for (int k0 = 0; k0 < K; k0 += 32) {
    const int buf = (k0 >> 5) & 1;
#if USE_TDM
    if (wave == 0) {
      int nxt = k0 + 32;
      if (nxt < K) {    // kick DMA for next tile, then wait for current tile
        tdm_load_tile((unsigned)(uintptr_t)&As[buf ^ 1][0][0],
                      A + (size_t)m0 * K + nxt, 128, K, M);
        tdm_load_tile((unsigned)(uintptr_t)&Bs[buf ^ 1][0][0],
                      W + (size_t)n0 * K + nxt, 128, K, N);
        __builtin_amdgcn_s_wait_tensorcnt(2);
      } else {
        __builtin_amdgcn_s_wait_tensorcnt(0);
      }
    }
    __syncthreads();
#else
    // manual staging: b128 global loads, b128 LDS stores
#pragma unroll
    for (int i = 0; i < 2; ++i) {
      int flat = i * 256 + tid;               // 512 quads for A (128 rows x 4)
      int r = flat >> 2, c = (flat & 3) * 4;
      *(u32x4v*)&As[buf][r][c] = *(const u32x4v*)(A + (size_t)(m0 + r) * K + k0 + c * 2);
    }
#pragma unroll
    for (int i = 0; i < 2; ++i) {
      int flat = i * 256 + tid;               // 512 quads for B (128 rows x 4)
      int r = flat >> 2, c = (flat & 3) * 4;
      *(u32x4v*)&Bs[buf][r][c] = *(const u32x4v*)(W + (size_t)(n0 + r) * K + k0 + c * 2);
    }
    __syncthreads();
#endif

    AFrag a[2], b[4];
#pragma unroll
    for (int t = 0; t < 2; ++t) {
      const int row = wm * 32 + t * 16 + lr;  // A frag: lane lr = row m
      a[t].q[0] = *(const u32x4v*)&As[buf][row][4 * half_];      // K=half*8..+7
      a[t].q[1] = *(const u32x4v*)&As[buf][row][8 + 4 * half_];  // K=16+half*8..
    }
#pragma unroll
    for (int t = 0; t < 4; ++t) {
      const int col = wn * 64 + t * 16 + lr;  // B frag: lane lr = col n
      b[t].q[0] = *(const u32x4v*)&Bs[buf][col][4 * half_];      // K=half*8..+7
      b[t].q[1] = *(const u32x4v*)&Bs[buf][col][8 + 4 * half_];
    }
#pragma unroll
    for (int tm = 0; tm < 2; ++tm)
#pragma unroll
      for (int tn = 0; tn < 4; ++tn)
        acc[tm][tn] = wmma32(a[tm], b[tn], acc[tm][tn]);
    __syncthreads();   // all frag reads done before DMA may overwrite this buf
  }

  // C layout: lane lr = column; rows m = half*8 + r live in VGPR r
#pragma unroll
  for (int tm = 0; tm < 2; ++tm)
#pragma unroll
    for (int tn = 0; tn < 4; ++tn)
#pragma unroll
      for (int r = 0; r < 8; ++r) {
        size_t row = (size_t)(m0 + wm * 32 + tm * 16 + half_ * 8 + r);
        C[row * N + n0 + wn * 64 + tn * 16 + lr] = acc[tm][tn][r];
      }
}

// ---------------- RoPE + transpose: (B,S,NH,HD) f32 -> (B,NH,S,HD) f16 ------
__global__ void rope_tr_kernel(const float* __restrict__ src, _Float16* __restrict__ dst,
                               const float* __restrict__ theta, int NH)
{
  size_t p = (size_t)blockIdx.x * 256 + threadIdx.x;  // one thread per pair
  int i = (int)(p & (HD_ / 2 - 1));
  size_t q = p >> 5;                                  // HD/2 == 32
  int h = (int)(q % NH); q /= NH;
  int s = (int)(q % S_);
  int b = (int)(q / S_);
  float th = theta[s * (HD_ / 2) + i];
  float c = __cosf(th), sn = __sinf(th);
  const float* sp = src + ((size_t)(b * S_ + s) * NH + h) * HD_ + 2 * i;
  float x0 = sp[0], x1 = sp[1];
  _Float16* dp = dst + ((size_t)(b * NH + h) * S_ + s) * HD_ + 2 * i;
  dp[0] = (_Float16)(x0 * c - x1 * sn);
  dp[1] = (_Float16)(x0 * sn + x1 * c);
}

// ---------------- V transpose: (B,S,KVH,HD) f32 -> (B,KVH,HD,S) f16 ---------
__global__ void v_tr_kernel(const float* __restrict__ src, _Float16* __restrict__ dst)
{
  size_t p = (size_t)blockIdx.x * 256 + threadIdx.x;
  int d = (int)(p & (HD_ - 1)); size_t q = p >> 6;
  int h = (int)(q & (KVH_ - 1)); q >>= 3;
  int s = (int)(q & (S_ - 1));
  int b = (int)(q >> 11);
  dst[((size_t)(b * KVH_ + h) * HD_ + d) * S_ + s] = (_Float16)src[p];
}

// ---------------- flash attention -------------------------------------------
// One wave per (b, h, 16-row q-tile). Q:(B,H,S,HD) K:(B,KVH,S,HD) Vt:(B,KVH,HD,S)
__global__ void __launch_bounds__(256) flash_kernel(
    const _Float16* __restrict__ Qh, const _Float16* __restrict__ Kh,
    const _Float16* __restrict__ Vt, _Float16* __restrict__ Oh)
{
  __shared__ __align__(16) _Float16 Pls[8][16][48]; // row stride 96B (16B mult)
  const int wave  = threadIdx.x >> 5;
  const int lane  = threadIdx.x & 31;
  const int half_ = lane >> 4;
  const int lr    = lane & 15;

  int task = blockIdx.x * 8 + wave;          // B*H*(S/16) = 8192 tasks
  int qt = task & 127; int t1 = task >> 7;
  int h  = t1 & 31;    int b  = t1 >> 5;
  int q0 = qt * 16;
  int kv = h >> 2;                           // GQA: 4 query heads per kv head

  const _Float16* qb  = Qh + ((size_t)(b * H_ + h) * S_ + q0) * HD_;
  const _Float16* kbp = Kh + ((size_t)(b * KVH_ + kv) * S_) * HD_;
  const _Float16* vbp = Vt + ((size_t)(b * KVH_ + kv) * HD_) * S_;

  // Q fragments, kept resident (two b128 loads each, ISA A-layout groups)
  AFrag aq[2];
#pragma unroll
  for (int kk = 0; kk < 2; ++kk) {
    const _Float16* p = qb + (size_t)lr * HD_ + kk * 32;
    aq[kk].q[0] = *(const u32x4v*)(p + 8 * half_);
    aq[kk].q[1] = *(const u32x4v*)(p + 16 + 8 * half_);
  }

  v8f accO[4] = {};
  float mi[8], li[8];
#pragma unroll
  for (int r = 0; r < 8; ++r) { mi[r] = -1e30f; li[r] = 0.0f; }

  for (int kb0 = 0; kb0 <= q0; kb0 += 32) {  // causal: kb0 <= q0 covers all live keys
    if (kb0 + 32 <= q0) {                    // prefetch next K/V tiles
      __builtin_prefetch(kbp + (size_t)(kb0 + 32 + lr) * HD_, 0, 1);
      __builtin_prefetch(vbp + (size_t)lr * S_ + kb0 + 32, 0, 1);
    }
    // scores: two 16(q) x 16(key) tiles
    v8f sc[2] = {};
#pragma unroll
    for (int nt = 0; nt < 2; ++nt) {
#pragma unroll
      for (int kk = 0; kk < 2; ++kk) {
        AFrag bk;   // B frag: two 8-f16 groups per lane (K=half*8.., 16+half*8..)
        const _Float16* p = kbp + (size_t)(kb0 + nt * 16 + lr) * HD_ + kk * 32;
        bk.q[0] = *(const u32x4v*)(p + 8 * half_);
        bk.q[1] = *(const u32x4v*)(p + 16 + 8 * half_);
        sc[nt] = wmma32(aq[kk], bk, sc[nt]);
      }
    }

    // online softmax per row (rows m = half*8 + r; columns striped across lanes)
#pragma unroll
    for (int r = 0; r < 8; ++r) {
      int qpos = q0 + half_ * 8 + r;
      float s0 = sc[0][r] * 0.125f;          // 1/sqrt(64)
      float s1 = sc[1][r] * 0.125f;
      if (kb0 + lr > qpos)      s0 = -1e30f; // causal mask
      if (kb0 + 16 + lr > qpos) s1 = -1e30f;
      float rm = fmaxf(s0, s1);
#pragma unroll
      for (int off = 8; off; off >>= 1)
        rm = fmaxf(rm, __shfl_xor(rm, off, 32));
      float mnew  = fmaxf(mi[r], rm);
      float alpha = __expf(mi[r] - mnew);
      float p0 = __expf(s0 - mnew);
      float p1 = __expf(s1 - mnew);
      float rs = p0 + p1;
#pragma unroll
      for (int off = 8; off; off >>= 1)
        rs += __shfl_xor(rs, off, 32);
      li[r] = li[r] * alpha + rs;
      mi[r] = mnew;
#pragma unroll
      for (int t = 0; t < 4; ++t) accO[t][r] = accO[t][r] * alpha;
      Pls[wave][half_ * 8 + r][lr]      = (_Float16)p0;
      Pls[wave][half_ * 8 + r][16 + lr] = (_Float16)p1;
    }
    // wave-private LDS round trip (C-layout -> A-layout); waves diverge, so no
    // workgroup barrier — order our own DS ops explicitly.
    asm volatile("s_wait_dscnt 0" ::: "memory");

    AFrag ap;
    ap.q[0] = *(const u32x4v*)&Pls[wave][lr][8 * half_];
    ap.q[1] = *(const u32x4v*)&Pls[wave][lr][16 + 8 * half_];

    // O += P * V  (4 hd-tiles of 16)
#pragma unroll
    for (int t = 0; t < 4; ++t) {
      AFrag bv;
      const _Float16* p = vbp + (size_t)(t * 16 + lr) * S_ + kb0;
      bv.q[0] = *(const u32x4v*)(p + 8 * half_);
      bv.q[1] = *(const u32x4v*)(p + 16 + 8 * half_);
      accO[t] = wmma32(ap, bv, accO[t]);
    }
  }

  // normalize and write Oh[(b,s),(h*HD+hd)]
#pragma unroll
  for (int t = 0; t < 4; ++t)
#pragma unroll
    for (int r = 0; r < 8; ++r) {
      float o = accO[t][r] / li[r];
      size_t row = (size_t)(b * S_ + q0 + half_ * 8 + r);
      Oh[(row * H_ + h) * HD_ + t * 16 + lr] = (_Float16)o;
    }
}

// ---------------- host side -------------------------------------------------
extern "C" void kernel_launch(void* const* d_in, const int* in_sizes, int n_in,
                              void* d_out, int out_size, void* d_ws, size_t ws_size,
                              hipStream_t stream)
{
  (void)in_sizes; (void)n_in; (void)out_size; (void)ws_size;
  const float* x     = (const float*)d_in[0];   // (B,S,D)
  const float* theta = (const float*)d_in[1];   // (S,HD/2)
  // d_in[2] = causal mask, recomputed analytically
  const float* wq    = (const float*)d_in[3];   // (H*HD, D)
  const float* wk    = (const float*)d_in[4];   // (KVH*HD, D)
  const float* wv    = (const float*)d_in[5];   // (KVH*HD, D)
  const float* wo    = (const float*)d_in[6];   // (D, H*HD)
  float* out = (float*)d_out;                   // (B,S,D) f32

  char* ws = (char*)d_ws;
  size_t off = 0;
  auto alloc = [&](size_t bytes) -> void* {
    void* p = ws + off;
    off += (bytes + 255) & ~(size_t)255;
    return p;
  };

  _Float16* xh  = (_Float16*)alloc((size_t)BS_ * D_ * 2);
  _Float16* wqh = (_Float16*)alloc((size_t)D_ * D_ * 2);
  _Float16* wkh = (_Float16*)alloc((size_t)KVD_ * D_ * 2);
  _Float16* wvh = (_Float16*)alloc((size_t)KVD_ * D_ * 2);
  _Float16* woh = (_Float16*)alloc((size_t)D_ * D_ * 2);
  float*    Qf  = (float*)   alloc((size_t)BS_ * D_ * 4);
  float*    Kf  = (float*)   alloc((size_t)BS_ * KVD_ * 4);
  float*    Vf  = (float*)   alloc((size_t)BS_ * KVD_ * 4);
  _Float16* Qh  = (_Float16*)alloc((size_t)BS_ * D_ * 2);    // (B,H,S,HD)
  _Float16* Kh  = (_Float16*)alloc((size_t)BS_ * KVD_ * 2);  // (B,KVH,S,HD)
  _Float16* Vth = (_Float16*)alloc((size_t)BS_ * KVD_ * 2);  // (B,KVH,HD,S)
  _Float16* Oh  = (_Float16*)alloc((size_t)BS_ * D_ * 2);    // (B,S,H*HD)

  // 1) fp32 -> f16
  cvt_kernel<<<(BS_ * D_ / 4) / 256, 256, 0, stream>>>(x,  xh,  BS_ * D_ / 4);
  cvt_kernel<<<(D_ * D_ / 4) / 256, 256, 0, stream>>>(wq, wqh, D_ * D_ / 4);
  cvt_kernel<<<(KVD_ * D_ / 4) / 256, 256, 0, stream>>>(wk, wkh, KVD_ * D_ / 4);
  cvt_kernel<<<(KVD_ * D_ / 4) / 256, 256, 0, stream>>>(wv, wvh, KVD_ * D_ / 4);
  cvt_kernel<<<(D_ * D_ / 4) / 256, 256, 0, stream>>>(wo, woh, D_ * D_ / 4);

  // 2) projections: Q = x wq^T, K = x wk^T, V = x wv^T
  gemm_f16_nt<<<dim3(D_ / 128,  BS_ / 128), 256, 0, stream>>>(xh, wqh, Qf, BS_, D_,  D_);
  gemm_f16_nt<<<dim3(KVD_ / 128, BS_ / 128), 256, 0, stream>>>(xh, wkh, Kf, BS_, KVD_, D_);
  gemm_f16_nt<<<dim3(KVD_ / 128, BS_ / 128), 256, 0, stream>>>(xh, wvh, Vf, BS_, KVD_, D_);

  // 3) RoPE + layout transforms
  rope_tr_kernel<<<(BS_ * H_ * (HD_ / 2)) / 256, 256, 0, stream>>>(Qf, Qh, theta, H_);
  rope_tr_kernel<<<(BS_ * KVH_ * (HD_ / 2)) / 256, 256, 0, stream>>>(Kf, Kh, theta, KVH_);
  v_tr_kernel<<<(BS_ * KVD_) / 256, 256, 0, stream>>>(Vf, Vth);

  // 4) flash attention: 8192 wave-tasks, 8 waves per block
  flash_kernel<<<(B_ * H_ * (S_ / 16)) / 8, 256, 0, stream>>>(Qh, Kh, Vth, Oh);

  // 5) output projection: out = Oh wo^T
  gemm_f16_nt<<<dim3(D_ / 128, BS_ / 128), 256, 0, stream>>>(Oh, woh, out, BS_, D_, D_);
}
